// ParticleNet_69363721830758
// MI455X (gfx1250) — matrix operand order repaired and checked by reading, not compiled, not run on hardware
//
#include <hip/hip_runtime.h>
#include <hip/hip_bf16.h>

// ---------------- problem constants ----------------
#define NUMG   512
#define NPG    128
#define NN     65536
#define KEDGE  8
#define EE     524288
#define INCH   16
#define HH     128
#define EPSV   1e-5f

// ---------------- WMMA vector types (CDNA5, wave32) ----------------
typedef __attribute__((ext_vector_type(16))) __bf16 v16bf;
typedef __attribute__((ext_vector_type(8)))  __bf16 v8bf;
typedef __attribute__((ext_vector_type(8)))  float  v8f;

__device__ __forceinline__ unsigned short f2bf(float f) {
  unsigned u = __float_as_uint(f);
  unsigned r = u + 0x7FFFu + ((u >> 16) & 1u);   // round to nearest even
  return (unsigned short)(r >> 16);
}
__device__ __forceinline__ float bf2f(unsigned short h) {
  return __uint_as_float(((unsigned)h) << 16);
}

// ---------------------------------------------------------------------------
// Weight pre-swizzle: fp32 [Kin, Nout] row-major -> bf16 fragments laid out so
// a wave can load a 32x16 B tile with two b128 loads per lane.
// Layout: dst[((kt*NT + nt)*32 + lane)*16 + i]  holds  W[kt*32 + (lane>=16?16:0) + i,
//                                                        nt*16 + (lane&15)]
// Rows k >= Kin are zero (used to pad K=16 -> 32 for the QKV projections).
// ---------------------------------------------------------------------------
__global__ void k_convw(const float* __restrict__ W, unsigned short* __restrict__ dst,
                        int Kin, int Kp, int Nout) {
  int idx = blockIdx.x * 256 + threadIdx.x;
  if (idx >= Kp * Nout) return;
  int i    = idx & 15;
  int lane = (idx >> 4) & 31;
  int fn   = idx >> 9;
  int NT   = Nout >> 4;
  int nt   = fn % NT;
  int kt   = fn / NT;
  int kk   = kt * 32 + ((lane >= 16) ? 16 : 0) + i;
  int n    = nt * 16 + (lane & 15);
  float v  = (kk < Kin) ? W[(long)kk * Nout + n] : 0.f;
  dst[idx] = f2bf(v);
}

// ---------------------------------------------------------------------------
// Generic fused WMMA GEMM:  Out = act( stage(In) @ W + bias )
//   mode 0: In = fp32 [R, Kin]  (zero-padded to Kp; optional per-channel affine)
//   mode 1: In = bf16 [R, Kp]   (optional per-channel affine == folded BatchNorm)
//   mode 2: edge-concat gather: In = fp32 node feats [NN,128]; row e stages
//           [ x[tgt(e)] , x[src(e)] - x[tgt(e)] ]  (Kp = 256, tgt(e) = e>>3)
//   act 0: none, 1: relu, 2: selu
//   outBf / outF: optional bf16 / fp32 [R, Nout] outputs
//   stats: optional [2*Nout] (sum, sumsq) accumulated for the next BatchNorm
// Block = 256 threads = 8 waves; covers a 64-row x 64-col tile.
// ---------------------------------------------------------------------------
__global__ __launch_bounds__(256) void k_gemm(
    const unsigned short* __restrict__ Wf,
    const float* __restrict__ bias,
    const void* __restrict__ inP,
    const int* __restrict__ srcIdx,
    const float* __restrict__ affA,
    const float* __restrict__ affB,
    unsigned short* __restrict__ outBf,
    float* __restrict__ outF,
    float* __restrict__ stats,
    int Kp, int kshift, int Kin, int Nout, int mode, int act) {
  __shared__ __align__(32) unsigned short sA[64 * 256];
  __shared__ float sSum[64], sSq[64];

  const int tid = threadIdx.x;
  const long r0 = (long)blockIdx.x * 64;

  if (tid < 64) { sSum[tid] = 0.f; sSq[tid] = 0.f; }

  // ---- stage A tile into LDS as bf16 (with gather / BN-affine folding) ----
  const int total = 64 << kshift;
  if (mode == 2) {
    const float* xf = (const float*)inP;
    for (int idx = tid; idx < total; idx += 256) {
      int r = idx >> 8, c = idx & 255;
      long e = r0 + r;
      int tg = (int)(e >> 3);           // tgt = repeat(arange(N), 8)
      int cc = c & 127;
      float xi = xf[(long)tg * 128 + cc];
      float vv;
      if (c < 128) vv = xi;
      else {
        int sj = srcIdx[e];
        vv = xf[(long)sj * 128 + cc] - xi;
      }
      sA[idx] = f2bf(vv);
    }
  } else if (mode == 1) {
    const unsigned short* xh = (const unsigned short*)inP;
    for (int idx = tid; idx < total; idx += 256) {
      int r = idx >> kshift, c = idx & (Kp - 1);
      float vv = bf2f(xh[(r0 + r) * Kp + c]);
      if (affA) vv = affA[c] * vv + affB[c];   // folded BatchNorm
      sA[idx] = f2bf(vv);
    }
  } else {
    const float* xf = (const float*)inP;
    for (int idx = tid; idx < total; idx += 256) {
      int r = idx >> kshift, c = idx & (Kp - 1);
      float vv = 0.f;
      if (c < Kin) {
        vv = xf[(r0 + r) * Kin + c];
        if (affA) vv = affA[c] * vv + affB[c];
      }
      sA[idx] = f2bf(vv);
    }
  }
  __syncthreads();

  // ---- WMMA tiles ----
  const int w    = tid >> 5, lane = tid & 31;
  const int mt   = w & 3;            // 4 M-tiles of 16 rows
  const int nh   = w >> 2;           // 2 N-halves of 32 cols
  const int NT   = Nout >> 4;
  const int nt0  = blockIdx.y * 4 + nh * 2;
  const int mrow = mt * 16 + (lane & 15);
  const int kbh  = (lane >= 16) ? 16 : 0;  // A half-wave K offset
  const int mup  = (lane >= 16) ? 8 : 0;   // C/D half-wave M offset

  v8f acc0 = {0.f,0.f,0.f,0.f,0.f,0.f,0.f,0.f};
  v8f acc1 = {0.f,0.f,0.f,0.f,0.f,0.f,0.f,0.f};

  const int KT = Kp >> 5;
  for (int kt = 0; kt < KT; ++kt) {
    union { v16bf v; v8bf h[2]; } a;
    const unsigned short* ap = &sA[mrow * Kp + kt * 32 + kbh];
    a.h[0] = *(const v8bf*)(ap);        // K +0..7
    a.h[1] = *(const v8bf*)(ap + 8);    // K +8..15
    const unsigned short* bp = Wf + ((long)(kt * NT + nt0) * 32 + lane) * 16;
    v16bf b0 = *(const v16bf*)(bp);
    v16bf b1 = *(const v16bf*)(bp + 512);
    acc0 = __builtin_amdgcn_wmma_f32_16x16x32_bf16(false, a.v, false, b0,
                                                   (short)0, acc0, false, false);
    acc1 = __builtin_amdgcn_wmma_f32_16x16x32_bf16(false, a.v, false, b1,
                                                   (short)0, acc1, false, false);
  }

  // ---- epilogue: bias + activation + store + BN stats ----
  for (int t = 0; t < 2; ++t) {
    v8f acc = t ? acc1 : acc0;
    int gn = (nt0 + t) * 16 + (lane & 15);
    float bi = bias ? bias[gn] : 0.f;
    float ls = 0.f, lq = 0.f;
    for (int r = 0; r < 8; ++r) {
      float vv = acc[r] + bi;
      if (act == 1)      vv = fmaxf(vv, 0.f);
      else if (act == 2) vv = (vv > 0.f) ? 1.0507009873554805f * vv
                 : 1.0507009873554805f * 1.6732632423543772f * (expf(vv) - 1.f);
      long row = r0 + mt * 16 + r + mup;
      if (outBf) outBf[row * Nout + gn] = f2bf(vv);
      if (outF)  outF [row * Nout + gn] = vv;
      ls += vv; lq += vv * vv;
    }
    if (stats) {
      int lc = nh * 32 + t * 16 + (lane & 15);
      atomicAdd(&sSum[lc], ls);
      atomicAdd(&sSq[lc], lq);
    }
  }
  if (stats) {
    __syncthreads();
    if (tid < 64) {
      int gc = blockIdx.y * 64 + tid;
      atomicAdd(&stats[gc],        sSum[tid]);
      atomicAdd(&stats[Nout + gc], sSq[tid]);
    }
  }
}

// ---------------------------------------------------------------------------
// BN stats -> folded affine (a*x + b)
// ---------------------------------------------------------------------------
__global__ void k_bn_finalize(const float* __restrict__ stats, float cnt,
                              const float* __restrict__ gamma,
                              const float* __restrict__ beta,
                              float* __restrict__ aA, float* __restrict__ aB, int C) {
  int c = blockIdx.x * blockDim.x + threadIdx.x;
  if (c >= C) return;
  float mean = stats[c] / cnt;
  float var  = stats[C + c] / cnt - mean * mean;
  float inv  = rsqrtf(var + EPSV);
  float a    = (gamma ? gamma[c] : 1.f) * inv;
  aA[c] = a;
  aB[c] = (beta ? beta[c] : 0.f) - mean * a;
}

// ---------------------------------------------------------------------------
// GraphNorm: one block per graph (128 nodes x 16 channels)
// ---------------------------------------------------------------------------
__global__ void k_graph_norm(const float* __restrict__ x, const float* __restrict__ w,
                             const float* __restrict__ b, const float* __restrict__ ms,
                             float* __restrict__ h0) {
  __shared__ float red[128];
  __shared__ float mean[16], varv[16];
  int g = blockIdx.x, t = threadIdx.x;
  int ch = t & 15, sub = t >> 4;
  long base = (long)g * NPG * INCH;
  float p = 0.f;
  for (int k = 0; k < 16; ++k) p += x[base + (sub + k * 8) * INCH + ch];
  red[t] = p; __syncthreads();
  if (t < 16) {
    float s = 0.f; for (int j = 0; j < 8; ++j) s += red[j * 16 + t];
    mean[t] = s * (1.f / NPG);
  }
  __syncthreads();
  float mm = mean[ch] * ms[ch];
  p = 0.f;
  for (int k = 0; k < 16; ++k) {
    float o = x[base + (sub + k * 8) * INCH + ch] - mm;
    p += o * o;
  }
  red[t] = p; __syncthreads();
  if (t < 16) {
    float s = 0.f; for (int j = 0; j < 8; ++j) s += red[j * 16 + t];
    varv[t] = s * (1.f / NPG);
  }
  __syncthreads();
  float inv = rsqrtf(varv[ch] + EPSV);
  float ww = w[ch], bb = b[ch];
  for (int k = 0; k < 16; ++k) {
    long idx = base + (sub + k * 8) * INCH + ch;
    h0[idx] = ww * (x[idx] - mm) * inv + bb;
  }
}

// ---------------------------------------------------------------------------
// TransformerConv attention: one wave per target node, 8 contiguous edges.
// q/k/v/s are bf16 [N,128]; output c1 = elu(attn + skip) fp32.
// ---------------------------------------------------------------------------
__global__ __launch_bounds__(256) void k_attention(
    const unsigned short* __restrict__ q, const unsigned short* __restrict__ kk,
    const unsigned short* __restrict__ v, const unsigned short* __restrict__ s,
    const int* __restrict__ src, float* __restrict__ c1) {
  int wv = threadIdx.x >> 5, lane = threadIdx.x & 31;
  long node = (long)blockIdx.x * 8 + wv;
  int c0 = lane * 4;
  long qb = node * HH + c0;
  float qf[4];
  for (int i = 0; i < 4; ++i) qf[i] = bf2f(q[qb + i]);
  int sj[8]; float logit[8];
  for (int j = 0; j < 8; ++j) {
    sj[j] = src[node * KEDGE + j];
    long kb = (long)sj[j] * HH + c0;
    float p = 0.f;
    for (int i = 0; i < 4; ++i) p += qf[i] * bf2f(kk[kb + i]);
    for (int m = 16; m >= 1; m >>= 1) p += __shfl_xor(p, m, 32);
    logit[j] = p * 0.0883883476483184f;   // 1/sqrt(128)
  }
  float mx = logit[0];
  for (int j = 1; j < 8; ++j) mx = fmaxf(mx, logit[j]);
  float ex[8], den = 0.f;
  for (int j = 0; j < 8; ++j) { ex[j] = expf(logit[j] - mx); den += ex[j]; }
  float inv = 1.f / den;
  float acc[4] = {0.f, 0.f, 0.f, 0.f};
  for (int j = 0; j < 8; ++j) {
    float a = ex[j] * inv;
    long vb = (long)sj[j] * HH + c0;
    for (int i = 0; i < 4; ++i) acc[i] += a * bf2f(v[vb + i]);
  }
  for (int i = 0; i < 4; ++i) {
    float o = acc[i] + bf2f(s[qb + i]);
    o = (o > 0.f) ? o : (expf(o) - 1.f);          // elu
    c1[qb + i] = o;
  }
}

// ---------------------------------------------------------------------------
// EdgeConv epilogue: per-node mean over 8 contiguous edges of bn3(h3)
// plus the relu shortcut.
// ---------------------------------------------------------------------------
__global__ void k_aggregate(const unsigned short* __restrict__ h3,
                            const float* __restrict__ aA, const float* __restrict__ aB,
                            const float* __restrict__ sc, float* __restrict__ cout) {
  long idx = (long)blockIdx.x * 256 + threadIdx.x;   // over N*128
  int c = (int)(idx & 127);
  long node = idx >> 7;
  long base = node * KEDGE * HH + c;
  float sum = 0.f;
  for (int j = 0; j < 8; ++j) sum += bf2f(h3[base + j * HH]);
  cout[idx] = aA[c] * (sum * 0.125f) + aB[c] + sc[idx];
}

// ---------------------------------------------------------------------------
// Global mean pool (h = c1+c2+c3) per graph + stats for bn0.
// ---------------------------------------------------------------------------
__global__ void k_pool(const float* __restrict__ c1, const float* __restrict__ c2,
                       const float* __restrict__ c3, float* __restrict__ g,
                       float* __restrict__ stats) {
  int gr = blockIdx.x, c = threadIdx.x;  // 128 threads
  float s = 0.f;
  for (int n = 0; n < NPG; ++n) {
    long idx = ((long)gr * NPG + n) * HH + c;
    s += c1[idx] + c2[idx] + c3[idx];
  }
  float m = s * (1.f / NPG);
  g[(long)gr * HH + c] = m;
  atomicAdd(&stats[c], m);
  atomicAdd(&stats[HH + c], m * m);
}

// ---------------------------------------------------------------------------
// Final 128->2 linear + log_softmax.
// ---------------------------------------------------------------------------
__global__ void k_out(const float* __restrict__ gd2, const float* __restrict__ W,
                      const float* __restrict__ b, float* __restrict__ out) {
  int r = blockIdx.x * blockDim.x + threadIdx.x;
  if (r >= NUMG) return;
  float o0 = b[0], o1 = b[1];
  for (int c = 0; c < HH; ++c) {
    float g = gd2[(long)r * HH + c];
    o0 += g * W[c * 2 + 0];
    o1 += g * W[c * 2 + 1];
  }
  float mx = fmaxf(o0, o1);
  float l = mx + logf(expf(o0 - mx) + expf(o1 - mx));
  out[r * 2 + 0] = o0 - l;
  out[r * 2 + 1] = o1 - l;
}

// ===========================================================================
extern "C" void kernel_launch(void* const* d_in, const int* in_sizes, int n_in,
                              void* d_out, int out_size, void* d_ws, size_t ws_size,
                              hipStream_t stream) {
  (void)in_sizes; (void)n_in; (void)out_size; (void)ws_size;

  // ---- inputs in setup_inputs() dict (insertion) order ----
  int ii = 0;
  const float* x     = (const float*)d_in[ii++];   // [N,16]
  const int*   eidx  = (const int*)  d_in[ii++];   // [2,E]
  ii++;                                            // batch (implicit layout)
  const float* gn_w  = (const float*)d_in[ii++];
  const float* gn_b  = (const float*)d_in[ii++];
  const float* gn_ms = (const float*)d_in[ii++];
  const float* bn0g  = (const float*)d_in[ii++];
  const float* bn0b  = (const float*)d_in[ii++];
  const float* Wq = (const float*)d_in[ii++]; const float* bq = (const float*)d_in[ii++];
  const float* Wk = (const float*)d_in[ii++]; const float* bk = (const float*)d_in[ii++];
  const float* Wv = (const float*)d_in[ii++]; const float* bv = (const float*)d_in[ii++];
  const float* Wsk = (const float*)d_in[ii++]; const float* bsk = (const float*)d_in[ii++];
  struct ECP { const float *l1W,*l1b,*l2W,*l2b,*l3W,*l3b,
                           *bn1g,*bn1b,*bn2g,*bn2b,*bn3g,*bn3b,*scW,*scb; };
  ECP ec[2];
  for (int e = 0; e < 2; ++e) {
    ec[e].l1W = (const float*)d_in[ii++]; ec[e].l1b = (const float*)d_in[ii++];
    ec[e].l2W = (const float*)d_in[ii++]; ec[e].l2b = (const float*)d_in[ii++];
    ec[e].l3W = (const float*)d_in[ii++]; ec[e].l3b = (const float*)d_in[ii++];
    ec[e].bn1g = (const float*)d_in[ii++]; ec[e].bn1b = (const float*)d_in[ii++];
    ec[e].bn2g = (const float*)d_in[ii++]; ec[e].bn2b = (const float*)d_in[ii++];
    ec[e].bn3g = (const float*)d_in[ii++]; ec[e].bn3b = (const float*)d_in[ii++];
    ec[e].scW = (const float*)d_in[ii++]; ec[e].scb = (const float*)d_in[ii++];
  }
  const float* d1W = (const float*)d_in[ii++]; const float* d1b = (const float*)d_in[ii++];
  const float* d2W = (const float*)d_in[ii++]; const float* d2b = (const float*)d_in[ii++];
  const float* oW  = (const float*)d_in[ii++]; const float* ob  = (const float*)d_in[ii++];
  const int* src = eidx;                      // edge_index[0] = src

  // ---- workspace layout ----
  char* ws = (char*)d_ws;
  size_t off = 0;
  auto alloc = [&](size_t bytes) -> char* {
    char* p = ws + off;
    off += (bytes + 255) & ~(size_t)255;
    return p;
  };
  unsigned short* wQ  = (unsigned short*)alloc(4096u * 2);
  unsigned short* wK  = (unsigned short*)alloc(4096u * 2);
  unsigned short* wV  = (unsigned short*)alloc(4096u * 2);
  unsigned short* wS  = (unsigned short*)alloc(4096u * 2);
  unsigned short* wEC[2][4];
  for (int e = 0; e < 2; ++e) {
    wEC[e][0] = (unsigned short*)alloc(32768u * 2);   // l1 [256,128]
    wEC[e][1] = (unsigned short*)alloc(16384u * 2);   // l2
    wEC[e][2] = (unsigned short*)alloc(16384u * 2);   // l3
    wEC[e][3] = (unsigned short*)alloc(16384u * 2);   // sc
  }
  unsigned short* wD1 = (unsigned short*)alloc(16384u * 2);
  unsigned short* wD2 = (unsigned short*)alloc(16384u * 2);

  float* stats = (float*)alloc(7u * 256 * 4);   // 6 edge BNs + bn0
  float* aff   = (float*)alloc(7u * 256 * 4);   // folded affines

  float*          h0  = (float*)alloc((size_t)NN * INCH * 4);
  unsigned short* qB  = (unsigned short*)alloc((size_t)NN * HH * 2);
  unsigned short* kB  = (unsigned short*)alloc((size_t)NN * HH * 2);
  unsigned short* vB  = (unsigned short*)alloc((size_t)NN * HH * 2);
  unsigned short* sB  = (unsigned short*)alloc((size_t)NN * HH * 2);
  float*          c1  = (float*)alloc((size_t)NN * HH * 4);
  float*          c2  = (float*)alloc((size_t)NN * HH * 4);
  float*          c3  = (float*)alloc((size_t)NN * HH * 4);
  float*          scF = (float*)alloc((size_t)NN * HH * 4);
  unsigned short* ehA = (unsigned short*)alloc((size_t)EE * HH * 2);
  unsigned short* ehB = (unsigned short*)alloc((size_t)EE * HH * 2);
  float*          g   = (float*)alloc((size_t)NUMG * HH * 4);
  unsigned short* gd1 = (unsigned short*)alloc((size_t)NUMG * HH * 2);
  float*          gd2 = (float*)alloc((size_t)NUMG * HH * 4);

  hipMemsetAsync(stats, 0, 7u * 256 * sizeof(float), stream);

  // ---- weight pre-swizzle ----
  auto convw = [&](const float* W, unsigned short* dst, int Kin, int Kp, int Nout) {
    int total = Kp * Nout;
    k_convw<<<dim3((total + 255) / 256), dim3(256), 0, stream>>>(W, dst, Kin, Kp, Nout);
  };
  convw(Wq, wQ, 16, 32, 128); convw(Wk, wK, 16, 32, 128);
  convw(Wv, wV, 16, 32, 128); convw(Wsk, wS, 16, 32, 128);
  for (int e = 0; e < 2; ++e) {
    convw(ec[e].l1W, wEC[e][0], 256, 256, 128);
    convw(ec[e].l2W, wEC[e][1], 128, 128, 128);
    convw(ec[e].l3W, wEC[e][2], 128, 128, 128);
    convw(ec[e].scW, wEC[e][3], 128, 128, 128);
  }
  convw(d1W, wD1, 128, 128, 128);
  convw(d2W, wD2, 128, 128, 128);

  auto gemm = [&](const unsigned short* Wf, const float* bias, const void* in,
                  const int* srcI, const float* aA, const float* aB,
                  unsigned short* oB, float* oF, float* st,
                  long R, int Kp, int ksh, int Kin, int Nout, int mode, int act) {
    dim3 grid((unsigned)(R / 64), (unsigned)(Nout / 64));
    k_gemm<<<grid, dim3(256), 0, stream>>>(Wf, bias, in, srcI, aA, aB, oB, oF, st,
                                           Kp, ksh, Kin, Nout, mode, act);
  };

  // ---- GraphNorm ----
  k_graph_norm<<<dim3(NUMG), dim3(128), 0, stream>>>(x, gn_w, gn_b, gn_ms, h0);

  // ---- TransformerConv ----
  gemm(wQ, bq,  h0, nullptr, nullptr, nullptr, qB, nullptr, nullptr, NN, 32, 5, 16, 128, 0, 0);
  gemm(wK, bk,  h0, nullptr, nullptr, nullptr, kB, nullptr, nullptr, NN, 32, 5, 16, 128, 0, 0);
  gemm(wV, bv,  h0, nullptr, nullptr, nullptr, vB, nullptr, nullptr, NN, 32, 5, 16, 128, 0, 0);
  gemm(wS, bsk, h0, nullptr, nullptr, nullptr, sB, nullptr, nullptr, NN, 32, 5, 16, 128, 0, 0);
  k_attention<<<dim3(NN / 8), dim3(256), 0, stream>>>(qB, kB, vB, sB, src, c1);

  // ---- EdgeConv x2 (BN folded into next layer's staging) ----
  auto edgeconv = [&](const float* cin, float* cout, int e, int statBase) {
    float* st1 = stats + (size_t)(statBase + 0) * 256;
    float* st2 = stats + (size_t)(statBase + 1) * 256;
    float* st3 = stats + (size_t)(statBase + 2) * 256;
    float* a1  = aff   + (size_t)(statBase + 0) * 256;
    float* a2  = aff   + (size_t)(statBase + 1) * 256;
    float* a3  = aff   + (size_t)(statBase + 2) * 256;
    gemm(wEC[e][0], ec[e].l1b, cin, src, nullptr, nullptr, ehA, nullptr, st1,
         EE, 256, 8, 256, 128, 2, 1);
    k_bn_finalize<<<dim3(1), dim3(128), 0, stream>>>(st1, (float)EE, ec[e].bn1g, ec[e].bn1b,
                                                     a1, a1 + 128, 128);
    gemm(wEC[e][1], ec[e].l2b, ehA, nullptr, a1, a1 + 128, ehB, nullptr, st2,
         EE, 128, 7, 128, 128, 1, 1);
    k_bn_finalize<<<dim3(1), dim3(128), 0, stream>>>(st2, (float)EE, ec[e].bn2g, ec[e].bn2b,
                                                     a2, a2 + 128, 128);
    gemm(wEC[e][2], ec[e].l3b, ehB, nullptr, a2, a2 + 128, ehA, nullptr, st3,
         EE, 128, 7, 128, 128, 1, 1);
    k_bn_finalize<<<dim3(1), dim3(128), 0, stream>>>(st3, (float)EE, ec[e].bn3g, ec[e].bn3b,
                                                     a3, a3 + 128, 128);
    gemm(wEC[e][3], ec[e].scb, cin, nullptr, nullptr, nullptr, nullptr, scF, nullptr,
         NN, 128, 7, 128, 128, 0, 1);
    k_aggregate<<<dim3((unsigned)((long)NN * HH / 256)), dim3(256), 0, stream>>>(
        ehA, a3, a3 + 128, scF, cout);
  };
  edgeconv(c1, c2, 0, 0);
  edgeconv(c2, c3, 1, 3);

  // ---- pool + bn0 + dense head ----
  float* st0 = stats + 6 * 256;
  float* a0  = aff   + 6 * 256;
  k_pool<<<dim3(NUMG), dim3(128), 0, stream>>>(c1, c2, c3, g, st0);
  k_bn_finalize<<<dim3(1), dim3(128), 0, stream>>>(st0, (float)NUMG, bn0g, bn0b,
                                                   a0, a0 + 128, 128);
  gemm(wD1, d1b, g,   nullptr, a0, a0 + 128, gd1, nullptr, nullptr, NUMG, 128, 7, 128, 128, 0, 2);
  gemm(wD2, d2b, gd1, nullptr, nullptr, nullptr, nullptr, gd2, nullptr, NUMG, 128, 7, 128, 128, 1, 2);
  k_out<<<dim3(2), dim3(256), 0, stream>>>(gd2, oW, ob, (float*)d_out);
}
